// GatedGCNNet1_57243324121154
// MI455X (gfx1250) — matrix operand order repaired and checked by reading, not compiled, not run on hardware
//
#include <hip/hip_runtime.h>
#include <math.h>

typedef __attribute__((ext_vector_type(16))) _Float16 v16h;
typedef __attribute__((ext_vector_type(8)))  float    v8f;

#define N_NODES 100000
#define N_EDGES 1000000
#define DIM     70
#define K_IN    64
#define N_GRAPH 100
#define COLT    5          // 5 column tiles of 16 (covers 70, padded to 80)
#define WPB     8          // waves per block (256 threads)
#define EPS_BN   1e-5f
#define EPS_GATE 1e-6f

// ---------------------------------------------------------------------------
// WMMA fragment builders (f32 source -> f16 fragments), per ISA 7.12.2 layouts
// A 16x32 f16: lane<16: row=r0+lane, halves {K 0..7, K 16..23}
//              lane>=16: row=r0+lane-16, halves {K 8..15, K 24..31}
// ---------------------------------------------------------------------------
__device__ __forceinline__ v16h load_a_f32(const float* __restrict__ X, int ld,
                                           int r0, int kc, int Klim, int Rlim) {
  int lane = threadIdx.x & 31;
  int r  = r0 + (lane & 15);
  int kb = kc + ((lane >> 4) << 3);  // +0 or +8
  v16h a;
  if (r < Rlim) {
    const float* row = X + (size_t)r * ld;
#pragma unroll
    for (int t = 0; t < 8; ++t) {
      int k0 = kb + t;
      int k1 = kb + 16 + t;
      a[t]     = (k0 < Klim) ? (_Float16)row[k0] : (_Float16)0.0f;
      a[t + 8] = (k1 < Klim) ? (_Float16)row[k1] : (_Float16)0.0f;
    }
  } else {
#pragma unroll
    for (int t = 0; t < 16; ++t) a[t] = (_Float16)0.0f;
  }
  return a;
}

// B 32x16 f16 (K x N, row-major weight W[k*ldw+j]):
// lane<16: col=j0+lane, K {kc+0..7, kc+16..23} ; lane>=16: col=j0+lane-16, K {kc+8..15, kc+24..31}
__device__ __forceinline__ v16h make_b_frag(const float* __restrict__ W, int ldw,
                                            int j0, int kc, int Klim, int Nlim, int lane) {
  int j  = j0 + (lane & 15);
  int kb = kc + ((lane >> 4) << 3);  // +0 or +8
  v16h b;
  if (j < Nlim) {
#pragma unroll
    for (int t = 0; t < 16; ++t) {
      int k = kb + t + ((t >> 3) << 3);
      b[t] = (k < Klim) ? (_Float16)W[(size_t)k * ldw + j] : (_Float16)0.0f;
    }
  } else {
#pragma unroll
    for (int t = 0; t < 16; ++t) b[t] = (_Float16)0.0f;
  }
  return b;
}

// ---------------------------------------------------------------------------
// Embedding GEMM: h = nodes_feat[100000,64] @ Wh[64,70] + bh
// 8 waves/block; W fragments staged to LDS once; 5 accumulators per wave.
// ---------------------------------------------------------------------------
__global__ __launch_bounds__(256)
void embed_h_kernel(const float* __restrict__ X, const float* __restrict__ W,
                    const float* __restrict__ bias, float* __restrict__ H) {
  __shared__ v16h Bs[2][COLT][32];              // 10 KB
  int lane = threadIdx.x & 31;
  int wv   = threadIdx.x >> 5;
  for (int f = wv; f < 2 * COLT; f += WPB)
    Bs[f / COLT][f % COLT][lane] = make_b_frag(W, DIM, (f % COLT) * 16,
                                               (f / COLT) * 32, K_IN, DIM, lane);
  __syncthreads();

  int r0 = (blockIdx.x * WPB + wv) * 16;
  {   // prefetch this wave's A rows (global_prefetch_b8)
    int pr = r0 + (lane & 15);
    if (pr >= N_NODES) pr = N_NODES - 1;
    __builtin_prefetch(X + (size_t)pr * K_IN, 0, 1);
  }
  v8f acc[COLT] = {};
#pragma unroll
  for (int kcI = 0; kcI < 2; ++kcI) {
    v16h a = load_a_f32(X, K_IN, r0, kcI * 32, K_IN, N_NODES);
#pragma unroll
    for (int j = 0; j < COLT; ++j) {
      v16h b = Bs[kcI][j][lane];
      acc[j] = __builtin_amdgcn_wmma_f32_16x16x32_f16(false, a, false, b, (short)0,
                                                      acc[j], false, false);
    }
  }
  int rbase = r0 + ((lane >> 4) << 3);
#pragma unroll
  for (int j = 0; j < COLT; ++j) {
    int col = j * 16 + (lane & 15);
    if (col < DIM) {
      float bj = bias[col];
#pragma unroll
      for (int v = 0; v < 8; ++v) {
        int r = rbase + v;
        if (r < N_NODES) H[(size_t)r * DIM + col] = acc[j][v] + bj;
      }
    }
  }
}

// e0[i,j] = edges_feat[i] * We[j] + be[j]
__global__ void embed_e_kernel(const float* __restrict__ ef, const float* __restrict__ We,
                               const float* __restrict__ be, float* __restrict__ E) {
  size_t idx = (size_t)blockIdx.x * blockDim.x + threadIdx.x;
  if (idx >= (size_t)N_EDGES * DIM) return;
  int j = (int)(idx % DIM);
  size_t i = idx / DIM;
  E[idx] = ef[i] * We[j] + be[j];
}

// ---------------------------------------------------------------------------
// Node projections Ah,Bh,Dh,Eh = h @ W[{0,1,3,4}] + b  (grid.y picks proj)
// ---------------------------------------------------------------------------
__global__ __launch_bounds__(256)
void node_proj_kernel(const float* __restrict__ H, const float* __restrict__ Wbase,
                      const float* __restrict__ bbase, float* __restrict__ out) {
  __shared__ v16h Bs[3][COLT][32];              // 15 KB
  int z = blockIdx.y;                 // 0..3
  int p = (z < 2) ? z : z + 1;        // weights 0,1,3,4 (A,B,D,E)
  const float* W    = Wbase + (size_t)p * DIM * DIM;
  const float* bias = bbase + (size_t)p * DIM;
  float* O = out + (size_t)z * N_NODES * DIM;

  int lane = threadIdx.x & 31;
  int wv   = threadIdx.x >> 5;
  for (int f = wv; f < 3 * COLT; f += WPB)
    Bs[f / COLT][f % COLT][lane] = make_b_frag(W, DIM, (f % COLT) * 16,
                                               (f / COLT) * 32, DIM, DIM, lane);
  __syncthreads();

  int r0 = (blockIdx.x * WPB + wv) * 16;
  {
    int pr = r0 + (lane & 15);
    if (pr >= N_NODES) pr = N_NODES - 1;
    __builtin_prefetch(H + (size_t)pr * DIM, 0, 1);
  }
  v8f acc[COLT] = {};
#pragma unroll
  for (int kcI = 0; kcI < 3; ++kcI) {
    v16h a = load_a_f32(H, DIM, r0, kcI * 32, DIM, N_NODES);
#pragma unroll
    for (int j = 0; j < COLT; ++j) {
      v16h b = Bs[kcI][j][lane];
      acc[j] = __builtin_amdgcn_wmma_f32_16x16x32_f16(false, a, false, b, (short)0,
                                                      acc[j], false, false);
    }
  }
  int rbase = r0 + ((lane >> 4) << 3);
#pragma unroll
  for (int j = 0; j < COLT; ++j) {
    int col = j * 16 + (lane & 15);
    if (col < DIM) {
      float bj = bias[col];
#pragma unroll
      for (int v = 0; v < 8; ++v) {
        int r = rbase + v;
        if (r < N_NODES) O[(size_t)r * DIM + col] = acc[j][v] + bj;
      }
    }
  }
}

// ---------------------------------------------------------------------------
// Edge GEMM fused with edge update: e_new = e@W_C + b_C + Dh[src] + Eh[dst]
// ---------------------------------------------------------------------------
__global__ __launch_bounds__(256)
void edge_gemm_kernel(const float* __restrict__ E, const float* __restrict__ W,
                      const float* __restrict__ bias,
                      const float* __restrict__ Dh, const float* __restrict__ Eh,
                      const int* __restrict__ src, const int* __restrict__ dst,
                      float* __restrict__ Enew) {
  __shared__ v16h Bs[3][COLT][32];              // 15 KB
  int lane = threadIdx.x & 31;
  int wv   = threadIdx.x >> 5;
  for (int f = wv; f < 3 * COLT; f += WPB)
    Bs[f / COLT][f % COLT][lane] = make_b_frag(W, DIM, (f % COLT) * 16,
                                               (f / COLT) * 32, DIM, DIM, lane);
  __syncthreads();

  int r0 = (blockIdx.x * WPB + wv) * 16;
  {
    int pr = r0 + (lane & 15);
    if (pr >= N_EDGES) pr = N_EDGES - 1;
    __builtin_prefetch(E + (size_t)pr * DIM, 0, 1);
  }
  v8f acc[COLT] = {};
#pragma unroll
  for (int kcI = 0; kcI < 3; ++kcI) {
    v16h a = load_a_f32(E, DIM, r0, kcI * 32, DIM, N_EDGES);
#pragma unroll
    for (int j = 0; j < COLT; ++j) {
      v16h b = Bs[kcI][j][lane];
      acc[j] = __builtin_amdgcn_wmma_f32_16x16x32_f16(false, a, false, b, (short)0,
                                                      acc[j], false, false);
    }
  }
  // hoist src/dst for this lane's 8 rows
  int rbase = r0 + ((lane >> 4) << 3);
  int sArr[8], dArr[8];
#pragma unroll
  for (int v = 0; v < 8; ++v) {
    int r = rbase + v;
    bool ok = r < N_EDGES;
    sArr[v] = ok ? src[r] : 0;
    dArr[v] = ok ? dst[r] : 0;
  }
#pragma unroll
  for (int j = 0; j < COLT; ++j) {
    int col = j * 16 + (lane & 15);
    if (col < DIM) {
      float bj = bias[col];
#pragma unroll
      for (int v = 0; v < 8; ++v) {
        int r = rbase + v;
        if (r < N_EDGES)
          Enew[(size_t)r * DIM + col] = acc[j][v] + bj +
              Dh[(size_t)sArr[v] * DIM + col] + Eh[(size_t)dArr[v] * DIM + col];
      }
    }
  }
}

// sigma = sigmoid(e_new); num[dst]+=sigma*Bh[src]; den[dst]+=sigma
__global__ void aggregate_kernel(const float* __restrict__ Enew, const float* __restrict__ Bh,
                                 const int* __restrict__ src, const int* __restrict__ dst,
                                 float* __restrict__ num, float* __restrict__ den) {
  size_t idx = (size_t)blockIdx.x * blockDim.x + threadIdx.x;
  if (idx >= (size_t)N_EDGES * DIM) return;
  int j = (int)(idx % DIM);
  size_t i = idx / DIM;
  float s = 1.0f / (1.0f + __expf(-Enew[idx]));
  int d = dst[i], sn = src[i];
  atomicAdd(&num[(size_t)d * DIM + j], s * Bh[(size_t)sn * DIM + j]);
  atomicAdd(&den[(size_t)d * DIM + j], s);
}

// h_new = (Ah + num/(den+eps)) * snorm_n
__global__ void hnew_kernel(const float* __restrict__ Ah, const float* __restrict__ num,
                            const float* __restrict__ den, const float* __restrict__ snorm_n,
                            float* __restrict__ Hn) {
  size_t idx = (size_t)blockIdx.x * blockDim.x + threadIdx.x;
  if (idx >= (size_t)N_NODES * DIM) return;
  size_t i = idx / DIM;
  Hn[idx] = (Ah[idx] + num[idx] / (den[idx] + EPS_GATE)) * snorm_n[i];
}

// BatchNorm pass 1: per-channel sum / sumsq (snorm optionally folded in)
__global__ void bn_stats_kernel(const float* __restrict__ X, const float* __restrict__ snorm,
                                long long R, float* __restrict__ sums) {
  __shared__ float ssum[DIM];
  __shared__ float ssq[DIM];
  for (int t = threadIdx.x; t < DIM; t += blockDim.x) { ssum[t] = 0.f; ssq[t] = 0.f; }
  __syncthreads();
  size_t total = (size_t)R * DIM;
  size_t stride = (size_t)gridDim.x * blockDim.x;
  for (size_t idx = (size_t)blockIdx.x * blockDim.x + threadIdx.x; idx < total; idx += stride) {
    int j = (int)(idx % DIM);
    size_t i = idx / DIM;
    float x = X[idx];
    if (snorm) x *= snorm[i];
    atomicAdd(&ssum[j], x);
    atomicAdd(&ssq[j], x * x);
  }
  __syncthreads();
  for (int t = threadIdx.x; t < DIM; t += blockDim.x) {
    atomicAdd(&sums[t], ssum[t]);
    atomicAdd(&sums[DIM + t], ssq[t]);
  }
}

// BatchNorm pass 2: fold into per-channel scale/shift
__global__ void bn_finalize_kernel(const float* __restrict__ sums, const float* __restrict__ gamma,
                                   const float* __restrict__ beta, float Rinv,
                                   float* __restrict__ ss) {
  int j = threadIdx.x;
  if (j < DIM) {
    float mu  = sums[j] * Rinv;
    float var = sums[DIM + j] * Rinv - mu * mu;
    float rs  = rsqrtf(var + EPS_BN);
    float sc  = rs * gamma[j];
    ss[j]       = sc;
    ss[DIM + j] = beta[j] - mu * sc;
  }
}

// Y += relu(x*scale + shift)  (residual in place; snorm optionally folded in)
__global__ void bn_apply_kernel(const float* __restrict__ X, const float* __restrict__ snorm,
                                const float* __restrict__ ss, float* __restrict__ Y,
                                long long R) {
  size_t idx = (size_t)blockIdx.x * blockDim.x + threadIdx.x;
  if (idx >= (size_t)R * DIM) return;
  int j = (int)(idx % DIM);
  size_t i = idx / DIM;
  float x = X[idx];
  if (snorm) x *= snorm[i];
  float y = x * ss[j] + ss[DIM + j];
  Y[idx] += (y > 0.f) ? y : 0.f;
}

// Per-graph mean readout
__global__ void readout_accum_kernel(const float* __restrict__ H, const int* __restrict__ gid,
                                     float* __restrict__ gsum, float* __restrict__ gcnt) {
  size_t idx = (size_t)blockIdx.x * blockDim.x + threadIdx.x;
  if (idx >= (size_t)N_NODES * DIM) return;
  int j = (int)(idx % DIM);
  size_t i = idx / DIM;
  int g = gid[i];
  atomicAdd(&gsum[(size_t)g * DIM + j], H[idx]);
  if (j == 0) atomicAdd(&gcnt[g], 1.0f);
}

__global__ void readout_final_kernel(const float* __restrict__ gsum, const float* __restrict__ gcnt,
                                     float* __restrict__ out) {
  int idx = blockIdx.x * blockDim.x + threadIdx.x;
  if (idx >= N_GRAPH * DIM) return;
  int g = idx / DIM;
  out[idx] = gsum[idx] / fmaxf(gcnt[g], 1.0f);
}

// ---------------------------------------------------------------------------
extern "C" void kernel_launch(void* const* d_in, const int* in_sizes, int n_in,
                              void* d_out, int out_size, void* d_ws, size_t ws_size,
                              hipStream_t stream) {
  (void)in_sizes; (void)n_in; (void)out_size; (void)ws_size;
  const float* nodes_feat = (const float*)d_in[0];
  const float* edges_feat = (const float*)d_in[1];
  const float* snorm_n    = (const float*)d_in[2];
  const float* snorm_e    = (const float*)d_in[3];
  const int*   src        = (const int*)d_in[4];
  const int*   dst        = (const int*)d_in[5];
  const int*   gid        = (const int*)d_in[6];
  const float* Wh         = (const float*)d_in[7];
  const float* bh         = (const float*)d_in[8];
  const float* We         = (const float*)d_in[9];
  const float* be         = (const float*)d_in[10];
  const float* W_gcn      = (const float*)d_in[11];
  const float* b_gcn      = (const float*)d_in[12];
  const float* gH         = (const float*)d_in[13];
  const float* bH         = (const float*)d_in[14];
  const float* gE         = (const float*)d_in[15];
  const float* bE         = (const float*)d_in[16];
  float* out = (float*)d_out;

  // workspace layout
  char* ws = (char*)d_ws;
  size_t off = 0;
  auto alloc = [&](size_t nfloats) -> float* {
    float* p = (float*)(ws + off);
    off += nfloats * sizeof(float);
    return p;
  };
  float* h     = alloc((size_t)N_NODES * DIM);      // node features
  float* proj  = alloc((size_t)4 * N_NODES * DIM);  // Ah,Bh,Dh,Eh
  float* e     = alloc((size_t)N_EDGES * DIM);      // edge features
  float* enew  = alloc((size_t)N_EDGES * DIM);      // raw e_new (pre-snorm)
  float* numb  = alloc((size_t)N_NODES * DIM);
  float* denb  = alloc((size_t)N_NODES * DIM);
  float* hnew  = alloc((size_t)N_NODES * DIM);
  float* stats = alloc(2 * DIM);
  float* ssbuf = alloc(2 * DIM);
  float* gsum  = alloc((size_t)N_GRAPH * DIM);
  float* gcnt  = alloc(N_GRAPH);

  const int ROWT_N = (N_NODES + 15) / 16;   // 6250 row tiles
  const int ROWT_E = (N_EDGES + 15) / 16;   // 62500 row tiles
  const int BLK_N  = (ROWT_N + WPB - 1) / WPB;   // 782
  const int BLK_E  = (ROWT_E + WPB - 1) / WPB;   // 7813
  const size_t ND = (size_t)N_NODES * DIM;
  const size_t ED = (size_t)N_EDGES * DIM;
  const int gN  = (int)((ND + 255) / 256);
  const int gE_ = (int)((ED + 255) / 256);

  // embeddings
  embed_h_kernel<<<BLK_N, 256, 0, stream>>>(nodes_feat, Wh, bh, h);
  embed_e_kernel<<<gE_, 256, 0, stream>>>(edges_feat, We, be, e);

  const int layers[2] = {0, 2};
  for (int L = 0; L < 2; ++L) {
    int li = layers[L];
    const float* Wb = W_gcn + (size_t)li * 5 * DIM * DIM;
    const float* bb = b_gcn + (size_t)li * 5 * DIM;

    // A,B,D,E node projections (WMMA)
    node_proj_kernel<<<dim3(BLK_N, 4), 256, 0, stream>>>(h, Wb, bb, proj);

    // Ce GEMM fused with Dh[src]+Eh[dst]+bias  (WMMA)
    edge_gemm_kernel<<<BLK_E, 256, 0, stream>>>(
        e, Wb + (size_t)2 * DIM * DIM, bb + (size_t)2 * DIM,
        proj + (size_t)2 * ND, proj + (size_t)3 * ND, src, dst, enew);

    // gated segment sums
    hipMemsetAsync(numb, 0, ND * sizeof(float), stream);
    hipMemsetAsync(denb, 0, ND * sizeof(float), stream);
    aggregate_kernel<<<gE_, 256, 0, stream>>>(enew, proj + (size_t)1 * ND, src, dst, numb, denb);

    // h_new = (Ah + num/(den+eps)) * snorm_n
    hnew_kernel<<<gN, 256, 0, stream>>>(proj, numb, denb, snorm_n, hnew);

    // node BN + relu + residual
    hipMemsetAsync(stats, 0, 2 * DIM * sizeof(float), stream);
    bn_stats_kernel<<<1024, 256, 0, stream>>>(hnew, nullptr, N_NODES, stats);
    bn_finalize_kernel<<<1, 128, 0, stream>>>(stats, gH + li * DIM, bH + li * DIM,
                                              1.0f / (float)N_NODES, ssbuf);
    bn_apply_kernel<<<gN, 256, 0, stream>>>(hnew, nullptr, ssbuf, h, N_NODES);

    // edge BN + relu + residual (snorm_e folded into stats and apply)
    hipMemsetAsync(stats, 0, 2 * DIM * sizeof(float), stream);
    bn_stats_kernel<<<2048, 256, 0, stream>>>(enew, snorm_e, N_EDGES, stats);
    bn_finalize_kernel<<<1, 128, 0, stream>>>(stats, gE + li * DIM, bE + li * DIM,
                                              1.0f / (float)N_EDGES, ssbuf);
    bn_apply_kernel<<<gE_, 256, 0, stream>>>(enew, snorm_e, ssbuf, e, N_EDGES);
  }

  // per-graph mean readout
  hipMemsetAsync(gsum, 0, (size_t)N_GRAPH * DIM * sizeof(float), stream);
  hipMemsetAsync(gcnt, 0, N_GRAPH * sizeof(float), stream);
  readout_accum_kernel<<<gN, 256, 0, stream>>>(h, gid, gsum, gcnt);
  readout_final_kernel<<<(N_GRAPH * DIM + 127) / 128, 128, 0, stream>>>(gsum, gcnt, out);
}
